// _MultiheadAttention2_70291434766604
// MI455X (gfx1250) — compile-verified, matrix operand order checked by reading, not run on hardware
//
#include <hip/hip_runtime.h>
#include <hip/hip_bf16.h>
#include <math.h>

typedef __attribute__((ext_vector_type(16))) _Float16 v16h;
typedef __attribute__((ext_vector_type(8)))  _Float16 v8h;
typedef __attribute__((ext_vector_type(8)))  float    v8f;
typedef __attribute__((ext_vector_type(4)))  float    v4f;

#define BSZ   2
#define LQ    2048
#define DIM   640
#define NH    8
#define CMAIN 512
#define AGGD  128
#define HD    64
#define MT    (BSZ*LQ)   // 4096

// ---------------------------------------------------------------------------
// Fragment loaders for v_wmma_f32_16x16x32_f16 (wave32).
// 16-bit A/B layout: lanes 0-15 hold K = {e, e+8 for e>=8}; lanes 16-31 add +8.
// So elements 0..7 are 8 contiguous halves at (k0 + khalf), elements 8..15 are
// 8 contiguous halves at (k0 + khalf + 16), khalf = (lane>=16) ? 8 : 0.
// ---------------------------------------------------------------------------
__device__ __forceinline__ v16h frag_from_f16(const _Float16* __restrict__ p) {
    v8h lo = *(const v8h*)(p);
    v8h hi = *(const v8h*)(p + 16);
    v16h r;
#pragma unroll
    for (int i = 0; i < 8; ++i) { r[i] = lo[i]; r[i + 8] = hi[i]; }
    return r;
}

__device__ __forceinline__ v16h frag_from_f32(const float* __restrict__ p) {
    v4f a0 = *(const v4f*)(p);
    v4f a1 = *(const v4f*)(p + 4);
    v4f a2 = *(const v4f*)(p + 16);
    v4f a3 = *(const v4f*)(p + 20);
    v16h r;
#pragma unroll
    for (int i = 0; i < 4; ++i) {
        r[i]      = (_Float16)a0[i];
        r[i + 4]  = (_Float16)a1[i];
        r[i + 8]  = (_Float16)a2[i];
        r[i + 12] = (_Float16)a3[i];
    }
    return r;
}

#define WMMA_F16(a, b, c) \
    __builtin_amdgcn_wmma_f32_16x16x32_f16(false, (a), false, (b), (short)0, (c), false, false)

// ---------------------------------------------------------------------------
// Transpose + f32->f16 convert a weight matrix W(rows,cols) -> WT[cols][rows]
// so WMMA B fragments (lane = output column, elements over K) load contiguously.
// ---------------------------------------------------------------------------
__global__ void wtrans_kernel(const float* __restrict__ W, _Float16* __restrict__ WT,
                              int rows, int cols) {
    int idx = blockIdx.x * 256 + threadIdx.x;
    if (idx < rows * cols) {
        int n = idx / rows;
        int k = idx % rows;
        WT[idx] = (_Float16)W[k * cols + n];
    }
}

// ---------------------------------------------------------------------------
// Fused QKV projection: (4096,640)x(640,640) per z in {q,k,v}.
// cols <  512 : add bias + W_pos, scatter to per-head f16 layouts
// cols >= 512 : f32 agg tail buffer (3,4096,128)
// ---------------------------------------------------------------------------
__global__ __launch_bounds__(128) void proj_kernel(
    const float* __restrict__ X,
    const _Float16* __restrict__ WQt, const _Float16* __restrict__ WKt,
    const _Float16* __restrict__ WVt,
    const float* __restrict__ bQ, const float* __restrict__ bK,
    const float* __restrict__ bV,
    const float* __restrict__ Wpos,
    _Float16* __restrict__ qh,   // (2,8,2048,64)
    _Float16* __restrict__ kh,   // (2,8,2048,64)
    _Float16* __restrict__ vhT,  // (2,8,64,2048)
    float* __restrict__ aggp)    // (3,4096,128)
{
    const int z = blockIdx.z;
    const _Float16* Wt = (z == 0) ? WQt : (z == 1) ? WKt : WVt;
    const float* bias  = (z == 0) ? bQ  : (z == 1) ? bK  : bV;

    const int lane  = threadIdx.x & 31;
    const int wave  = threadIdx.x >> 5;
    const int m0    = blockIdx.x * 64 + wave * 16;
    const int n0    = blockIdx.y * 64;
    const int khalf = (lane >> 4) * 8;
    const int arow  = m0 + (lane & 15);

    v8f acc[4] = {};
    for (int kk = 0; kk < DIM; kk += 32) {
        v16h a = frag_from_f32(X + arow * DIM + kk + khalf);
#pragma unroll
        for (int t = 0; t < 4; ++t) {
            int n = n0 + t * 16 + (lane & 15);
            v16h b = frag_from_f16(Wt + n * DIM + kk + khalf);
            acc[t] = WMMA_F16(a, b, acc[t]);
        }
    }

    const int rhi = (lane >= 16) ? 8 : 0;
#pragma unroll
    for (int t = 0; t < 4; ++t) {
#pragma unroll
        for (int r = 0; r < 8; ++r) {
            int m   = m0 + r + rhi;
            int col = n0 + t * 16 + (lane & 15);
            float v = acc[t][r] + bias[col];
            int b = m >> 11;
            int l = m & (LQ - 1);
            if (col < CMAIN) {
                v += Wpos[l * CMAIN + col];
                int h = col >> 6, d = col & 63;
                _Float16 hv = (_Float16)v;
                if (z == 0)      qh[(((b * NH + h) * LQ) + l) * HD + d] = hv;
                else if (z == 1) kh[(((b * NH + h) * LQ) + l) * HD + d] = hv;
                else             vhT[(((b * NH + h) * HD) + d) * LQ + l] = hv;
            } else {
                aggp[((size_t)z * MT + m) * AGGD + (col - CMAIN)] = v;
            }
        }
    }
}

// ---------------------------------------------------------------------------
// x_agg0 = gelu(mean(q,k,v tails) @ W_agg + b_agg) -> concat cols [512,640)
// ---------------------------------------------------------------------------
__global__ __launch_bounds__(128) void agg_kernel(
    const float* __restrict__ aggp, const _Float16* __restrict__ WAt,
    const float* __restrict__ bA, _Float16* __restrict__ concat)
{
    const int lane  = threadIdx.x & 31;
    const int wave  = threadIdx.x >> 5;
    const int m0    = blockIdx.x * 64 + wave * 16;
    const int n0    = blockIdx.y * 64;
    const int khalf = (lane >> 4) * 8;
    const int arow  = m0 + (lane & 15);

    v8f acc[4] = {};
    for (int kk = 0; kk < AGGD; kk += 32) {
        const float* p0 = aggp + (size_t)arow * AGGD + kk + khalf;
        const float* p1 = p0 + (size_t)MT * AGGD;
        const float* p2 = p1 + (size_t)MT * AGGD;
        v16h a;
#pragma unroll
        for (int i = 0; i < 8; ++i) {
            a[i]     = (_Float16)((p0[i] + p1[i] + p2[i]) * (1.0f / 3.0f));
            a[i + 8] = (_Float16)((p0[i + 16] + p1[i + 16] + p2[i + 16]) * (1.0f / 3.0f));
        }
#pragma unroll
        for (int t = 0; t < 4; ++t) {
            int n = n0 + t * 16 + (lane & 15);
            v16h b = frag_from_f16(WAt + n * AGGD + kk + khalf);
            acc[t] = WMMA_F16(a, b, acc[t]);
        }
    }

    const int rhi = (lane >= 16) ? 8 : 0;
#pragma unroll
    for (int t = 0; t < 4; ++t) {
#pragma unroll
        for (int r = 0; r < 8; ++r) {
            int m   = m0 + r + rhi;
            int col = n0 + t * 16 + (lane & 15);
            float x = acc[t][r] + bA[col];
            // jax.nn.gelu default (tanh approximation)
            float g = 0.5f * x * (1.0f + tanhf(0.7978845608028654f * (x + 0.044715f * x * x * x)));
            concat[(size_t)m * DIM + CMAIN + col] = (_Float16)g;
        }
    }
}

// ---------------------------------------------------------------------------
// scores[b,h] = (q_s @ k_s^T) * 1/sqrt(80), written f32 to the attn output.
// ---------------------------------------------------------------------------
__global__ __launch_bounds__(128) void scores_kernel(
    const _Float16* __restrict__ qh, const _Float16* __restrict__ kh,
    float* __restrict__ attn)
{
    const int bh    = blockIdx.z;
    const int lane  = threadIdx.x & 31;
    const int wave  = threadIdx.x >> 5;
    const int q0    = blockIdx.x * 64 + wave * 16;
    const int n0    = blockIdx.y * 64;
    const int khalf = (lane >> 4) * 8;

    const _Float16* qb = qh + (size_t)bh * LQ * HD;
    const _Float16* kb = kh + (size_t)bh * LQ * HD;

    v8f acc[4] = {};
#pragma unroll
    for (int kk = 0; kk < HD; kk += 32) {
        v16h a = frag_from_f16(qb + (q0 + (lane & 15)) * HD + kk + khalf);
#pragma unroll
        for (int t = 0; t < 4; ++t) {
            v16h b = frag_from_f16(kb + (n0 + t * 16 + (lane & 15)) * HD + kk + khalf);
            acc[t] = WMMA_F16(a, b, acc[t]);
        }
    }

    const float scale = 0.11180339887498949f; // (DIM/NHEADS)^-0.5 = 80^-0.5
    const int rhi = (lane >= 16) ? 8 : 0;
    float* ab = attn + (size_t)bh * LQ * LQ;
#pragma unroll
    for (int t = 0; t < 4; ++t) {
#pragma unroll
        for (int r = 0; r < 8; ++r) {
            ab[(size_t)(q0 + r + rhi) * LQ + n0 + t * 16 + (lane & 15)] = acc[t][r] * scale;
        }
    }
}

// ---------------------------------------------------------------------------
// In-place row softmax over 2048-wide rows. One 256-thread (8-wave) block/row.
// ---------------------------------------------------------------------------
__global__ __launch_bounds__(256) void softmax_kernel(float* __restrict__ attn)
{
    __shared__ float red[8];
    float* p = attn + (size_t)blockIdx.x * LQ;

    float v[8];
    float m = -3.0e38f;
#pragma unroll
    for (int i = 0; i < 8; ++i) { v[i] = p[threadIdx.x + i * 256]; m = fmaxf(m, v[i]); }
#pragma unroll
    for (int o = 16; o > 0; o >>= 1) m = fmaxf(m, __shfl_xor(m, o, 32));
    if ((threadIdx.x & 31) == 0) red[threadIdx.x >> 5] = m;
    __syncthreads();
    m = red[0];
#pragma unroll
    for (int i = 1; i < 8; ++i) m = fmaxf(m, red[i]);
    __syncthreads();

    float s = 0.0f;
#pragma unroll
    for (int i = 0; i < 8; ++i) { v[i] = __expf(v[i] - m); s += v[i]; }
#pragma unroll
    for (int o = 16; o > 0; o >>= 1) s += __shfl_xor(s, o, 32);
    if ((threadIdx.x & 31) == 0) red[threadIdx.x >> 5] = s;
    __syncthreads();
    s = 0.0f;
#pragma unroll
    for (int i = 0; i < 8; ++i) s += red[i];
    float inv = 1.0f / s;
#pragma unroll
    for (int i = 0; i < 8; ++i) p[threadIdx.x + i * 256] = v[i] * inv;
}

// ---------------------------------------------------------------------------
// out_head = attn @ v_s  -> concat cols [h*64, h*64+64) as f16.
// A (attn) converted f32->f16 at load; B from transposed V (contiguous in K).
// ---------------------------------------------------------------------------
__global__ __launch_bounds__(128) void av_kernel(
    const float* __restrict__ attn, const _Float16* __restrict__ vhT,
    _Float16* __restrict__ concat)
{
    const int bh    = blockIdx.y;
    const int lane  = threadIdx.x & 31;
    const int wave  = threadIdx.x >> 5;
    const int q0    = blockIdx.x * 64 + wave * 16;
    const int khalf = (lane >> 4) * 8;

    const float*    ab = attn + (size_t)bh * LQ * LQ + (size_t)(q0 + (lane & 15)) * LQ;
    const _Float16* vb = vhT + (size_t)bh * HD * LQ;

    v8f acc[4] = {};
    for (int kk = 0; kk < LQ; kk += 32) {
        __builtin_prefetch(ab + kk + 128, 0, 1);   // global_prefetch ahead on the A stream
        v16h a = frag_from_f32(ab + kk + khalf);
#pragma unroll
        for (int t = 0; t < 4; ++t) {
            v16h b = frag_from_f16(vb + (size_t)(t * 16 + (lane & 15)) * LQ + kk + khalf);
            acc[t] = WMMA_F16(a, b, acc[t]);
        }
    }

    const int rhi = (lane >= 16) ? 8 : 0;
    const int b = bh >> 3, h = bh & 7;
#pragma unroll
    for (int t = 0; t < 4; ++t) {
#pragma unroll
        for (int r = 0; r < 8; ++r) {
            int q = q0 + r + rhi;
            int d = t * 16 + (lane & 15);
            concat[(size_t)(b * LQ + q) * DIM + h * HD + d] = (_Float16)acc[t][r];
        }
    }
}

// ---------------------------------------------------------------------------
// Final projection: concat(4096,640) f16 @ W_out + b_out -> out f32.
// ---------------------------------------------------------------------------
__global__ __launch_bounds__(128) void outproj_kernel(
    const _Float16* __restrict__ concat, const _Float16* __restrict__ WOt,
    const float* __restrict__ bO, float* __restrict__ out)
{
    const int lane  = threadIdx.x & 31;
    const int wave  = threadIdx.x >> 5;
    const int m0    = blockIdx.x * 64 + wave * 16;
    const int n0    = blockIdx.y * 64;
    const int khalf = (lane >> 4) * 8;

    const _Float16* arow = concat + (size_t)(m0 + (lane & 15)) * DIM;

    v8f acc[4] = {};
    for (int kk = 0; kk < DIM; kk += 32) {
        __builtin_prefetch(arow + kk + 128, 0, 1);
        v16h a = frag_from_f16(arow + kk + khalf);
#pragma unroll
        for (int t = 0; t < 4; ++t) {
            int n = n0 + t * 16 + (lane & 15);
            v16h b = frag_from_f16(WOt + (size_t)n * DIM + kk + khalf);
            acc[t] = WMMA_F16(a, b, acc[t]);
        }
    }

    const int rhi = (lane >= 16) ? 8 : 0;
#pragma unroll
    for (int t = 0; t < 4; ++t) {
#pragma unroll
        for (int r = 0; r < 8; ++r) {
            int m   = m0 + r + rhi;
            int col = n0 + t * 16 + (lane & 15);
            out[(size_t)m * DIM + col] = acc[t][r] + bO[col];
        }
    }
}

// ---------------------------------------------------------------------------
extern "C" void kernel_launch(void* const* d_in, const int* in_sizes, int n_in,
                              void* d_out, int out_size, void* d_ws, size_t ws_size,
                              hipStream_t stream)
{
    const float* Q     = (const float*)d_in[0];
    const float* W_Q   = (const float*)d_in[1];
    const float* b_Q   = (const float*)d_in[2];
    const float* W_K   = (const float*)d_in[3];
    const float* b_K   = (const float*)d_in[4];
    const float* W_V   = (const float*)d_in[5];
    const float* b_V   = (const float*)d_in[6];
    const float* W_pos = (const float*)d_in[7];
    const float* W_agg = (const float*)d_in[8];
    const float* b_agg = (const float*)d_in[9];
    const float* W_out = (const float*)d_in[10];
    const float* b_out = (const float*)d_in[11];

    float* out  = (float*)d_out;                       // (2,2048,640)
    float* attn = out + (size_t)BSZ * LQ * DIM;        // (2,8,2048,2048)

    // Workspace carve-out (256B-aligned slices), ~27 MB total.
    char*  ws  = (char*)d_ws;
    size_t off = 0;
    auto carve = [&](size_t bytes) -> char* {
        char* p = ws + off;
        off += (bytes + 255) & ~(size_t)255;
        return p;
    };
    _Float16* WQt    = (_Float16*)carve((size_t)DIM * DIM * sizeof(_Float16));
    _Float16* WKt    = (_Float16*)carve((size_t)DIM * DIM * sizeof(_Float16));
    _Float16* WVt    = (_Float16*)carve((size_t)DIM * DIM * sizeof(_Float16));
    _Float16* WOt    = (_Float16*)carve((size_t)DIM * DIM * sizeof(_Float16));
    _Float16* WAt    = (_Float16*)carve((size_t)AGGD * AGGD * sizeof(_Float16));
    _Float16* qh     = (_Float16*)carve((size_t)BSZ * NH * LQ * HD * sizeof(_Float16));
    _Float16* kh     = (_Float16*)carve((size_t)BSZ * NH * LQ * HD * sizeof(_Float16));
    _Float16* vhT    = (_Float16*)carve((size_t)BSZ * NH * HD * LQ * sizeof(_Float16));
    float*    aggp   = (float*)   carve((size_t)3 * MT * AGGD * sizeof(float));
    _Float16* concat = (_Float16*)carve((size_t)MT * DIM * sizeof(_Float16));

    // 1) Transpose+convert weights to f16 [n][k] layouts.
    {
        int n = DIM * DIM;
        wtrans_kernel<<<(n + 255) / 256, 256, 0, stream>>>(W_Q, WQt, DIM, DIM);
        wtrans_kernel<<<(n + 255) / 256, 256, 0, stream>>>(W_K, WKt, DIM, DIM);
        wtrans_kernel<<<(n + 255) / 256, 256, 0, stream>>>(W_V, WVt, DIM, DIM);
        wtrans_kernel<<<(n + 255) / 256, 256, 0, stream>>>(W_out, WOt, DIM, DIM);
        int na = AGGD * AGGD;
        wtrans_kernel<<<(na + 255) / 256, 256, 0, stream>>>(W_agg, WAt, AGGD, AGGD);
    }

    // 2) Fused QKV projection (+bias +W_pos, head scatter, agg tail).
    proj_kernel<<<dim3(MT / 64, DIM / 64, 3), 128, 0, stream>>>(
        Q, WQt, WKt, WVt, b_Q, b_K, b_V, W_pos, qh, kh, vhT, aggp);

    // 3) Agg branch GEMM + GELU into concat[:, 512:640].
    agg_kernel<<<dim3(MT / 64, AGGD / 64), 128, 0, stream>>>(aggp, WAt, b_agg, concat);

    // 4) Attention scores (scaled) written straight to the attn output region.
    scores_kernel<<<dim3(LQ / 64, LQ / 64, BSZ * NH), 128, 0, stream>>>(qh, kh, attn);

    // 5) In-place row softmax.
    softmax_kernel<<<BSZ * NH * LQ, 256, 0, stream>>>(attn);

    // 6) attn @ V -> concat[:, :512].
    av_kernel<<<dim3(LQ / 64, BSZ * NH), 128, 0, stream>>>(attn, vhT, concat);

    // 7) Final projection + bias -> out.
    outproj_kernel<<<dim3(MT / 64, DIM / 64), 128, 0, stream>>>(concat, WOt, b_out, out);
}
